// STRPN_78743930404875
// MI455X (gfx1250) — matrix operand order repaired and checked by reading, not compiled, not run on hardware
//
#include <hip/hip_runtime.h>
#include <hip/hip_bf16.h>
#include <math.h>

typedef __attribute__((ext_vector_type(16))) _Float16 v16h;
typedef __attribute__((ext_vector_type(8)))  _Float16 h8;
typedef __attribute__((ext_vector_type(8)))  float    v8f;

#define IMG_H 50
#define IMG_W 75
#define SPAT  (IMG_H * IMG_W)          // 3750
#define CIN   1024
#define CR    512
#define N2    112                      // padded head channels (108 real)
#define N2R   108
#define NPROP 33750                    // SPAT * 9
#define POST  300
#define POOLE (300 * 1024 * 49)        // 15,052,800

// ---------------------------------------------------------------------------
// Prep kernels: dtype/layout conversion
// ---------------------------------------------------------------------------
__global__ void prep_input(const float* __restrict__ head, _Float16* __restrict__ inF) {
    int i = blockIdx.x * 256 + threadIdx.x;          // i = s*CIN + c
    if (i >= SPAT * CIN) return;
    int c = i % CIN;
    int s = i / CIN;
    inF[i] = (_Float16)head[(size_t)c * SPAT + s];   // NCHW -> NHWC f16
}

__global__ void prep_wrpn(const float* __restrict__ w, _Float16* __restrict__ wT) {
    int i = blockIdx.x * 256 + threadIdx.x;          // i = (tap*CIN + ci)*CR + co
    if (i >= 9 * CIN * CR) return;
    int co  = i % CR;
    int ci  = (i / CR) % CIN;
    int tap = i / (CR * CIN);
    wT[i] = (_Float16)w[((size_t)co * CIN + ci) * 9 + tap];
}

__global__ void prep_w2(const float* __restrict__ wc, const float* __restrict__ wb,
                        const float* __restrict__ wt, const float* __restrict__ bc,
                        const float* __restrict__ bb, const float* __restrict__ bt,
                        _Float16* __restrict__ w2, float* __restrict__ b2) {
    int i = blockIdx.x * 256 + threadIdx.x;          // i = k*N2 + n
    if (i >= CR * N2) return;
    int n = i % N2;
    int k = i / N2;
    float v = 0.f;
    if      (n < 18)  v = wc[(size_t)n * CR + k];
    else if (n < 54)  v = wb[(size_t)(n - 18) * CR + k];
    else if (n < 108) v = wt[(size_t)(n - 54) * CR + k];
    w2[i] = (_Float16)v;
    if (i < N2) {
        float bv = 0.f;
        if      (i < 18)  bv = bc[i];
        else if (i < 54)  bv = bb[i - 18];
        else if (i < 108) bv = bt[i - 54];
        b2[i] = bv;
    }
}

// ---------------------------------------------------------------------------
// GEMM 1: 3x3 conv as 9 shifted GEMMs. M=3750 N=512 K=1024 per tap.
// Block = 256 threads (8 waves), tile 128x64, wave tile 16x64 (4 WMMA accs).
// ---------------------------------------------------------------------------
__global__ __launch_bounds__(256) void rpn_conv_gemm(
    const _Float16* __restrict__ inF,   // [SPAT][CIN]
    const _Float16* __restrict__ wT,    // [9][CIN][CR]
    const float*    __restrict__ bias,  // [CR]
    _Float16*       __restrict__ rpnF)  // [SPAT][CR]
{
    __shared__ __align__(16) _Float16 As[128][40];   // +8 halfs pad
    __shared__ __align__(16) _Float16 Bs[64][40];    // transposed [n][k]

    const int t    = threadIdx.x;
    const int wave = t >> 5;
    const int lane = t & 31;
    const int m0   = blockIdx.x * 128;
    const int n0   = blockIdx.y * 64;

    v8f acc[4] = {};

    for (int tap = 0; tap < 9; ++tap) {
        const int dy = tap / 3 - 1;
        const int dx = tap % 3 - 1;
        for (int k0 = 0; k0 < CIN; k0 += 32) {
            // --- stage A tile: 128 rows x 32 halfs ---
            {
                int row = t >> 1;
                int kh  = (t & 1) * 16;
                int s   = m0 + row;
                int y   = s / IMG_W, x = s - y * IMG_W;
                int yy  = y + dy,    xx = x + dx;
                bool ok = (s < SPAT) && (yy >= 0) && (yy < IMG_H) && (xx >= 0) && (xx < IMG_W);
                uint4 v0 = {0u,0u,0u,0u}, v1 = {0u,0u,0u,0u};
                if (ok) {
                    const uint4* p = (const uint4*)(inF + (size_t)(yy * IMG_W + xx) * CIN + k0 + kh);
                    v0 = p[0]; v1 = p[1];
                }
                *(uint4*)&As[row][kh]     = v0;
                *(uint4*)&As[row][kh + 8] = v1;
            }
            // --- stage B tile transposed: 32 k x 64 n -> Bs[n][k] ---
            {
                int k  = t >> 3;
                int n8 = (t & 7) * 8;
                const _Float16* src = wT + ((size_t)tap * CIN + k0 + k) * CR + n0 + n8;
                h8 v = *(const h8*)src;
                #pragma unroll
                for (int j = 0; j < 8; ++j) Bs[n8 + j][k] = v[j];
            }
            __syncthreads();

            // --- fragments + WMMA ---
            int mrow = wave * 16 + (lane & 15);
            int kb   = (lane >> 4) * 8;
            h8 alo = *(const h8*)&As[mrow][kb];
            h8 ahi = *(const h8*)&As[mrow][16 + kb];
            v16h a;
            #pragma unroll
            for (int i = 0; i < 8; ++i) { a[i] = alo[i]; a[i + 8] = ahi[i]; }
            int kb2 = (lane >> 4) * 8;
            #pragma unroll
            for (int nt = 0; nt < 4; ++nt) {
                int ncol = nt * 16 + (lane & 15);
                h8 blo = *(const h8*)&Bs[ncol][kb2];
                h8 bhi = *(const h8*)&Bs[ncol][16 + kb2];
                v16h b;
                #pragma unroll
                for (int i = 0; i < 8; ++i) { b[i] = blo[i]; b[i + 8] = bhi[i]; }
                acc[nt] = __builtin_amdgcn_wmma_f32_16x16x32_f16(
                    false, a, false, b, (short)0, acc[nt], false, false);
            }
            __syncthreads();
        }
    }

    // --- bias + ReLU, store f16 ---
    const int lm = lane & 15;
    const int hi = lane >> 4;
    #pragma unroll
    for (int nt = 0; nt < 4; ++nt) {
        int n = n0 + nt * 16 + lm;
        float bv = bias[n];
        #pragma unroll
        for (int r = 0; r < 8; ++r) {
            int m = m0 + wave * 16 + r + hi * 8;
            if (m < SPAT) {
                float v = acc[nt][r] + bv;
                v = v > 0.f ? v : 0.f;
                rpnF[(size_t)m * CR + n] = (_Float16)v;
            }
        }
    }
}

// ---------------------------------------------------------------------------
// GEMM 2: fused 1x1 convs. M=3750 N=112(108) K=512. Wave tile 16x112.
// ---------------------------------------------------------------------------
__global__ __launch_bounds__(256) void head_gemm(
    const _Float16* __restrict__ rpnF,  // [SPAT][CR]
    const _Float16* __restrict__ w2,    // [CR][N2]
    const float*    __restrict__ b2,    // [N2]
    float*          __restrict__ heads) // [SPAT][N2R]
{
    __shared__ __align__(16) _Float16 As[128][40];
    __shared__ __align__(16) _Float16 Bs[N2][40];

    const int t    = threadIdx.x;
    const int wave = t >> 5;
    const int lane = t & 31;
    const int m0   = blockIdx.x * 128;

    v8f acc[7] = {};

    for (int k0 = 0; k0 < CR; k0 += 32) {
        {
            int row = t >> 1;
            int kh  = (t & 1) * 16;
            int s   = m0 + row;
            uint4 v0 = {0u,0u,0u,0u}, v1 = {0u,0u,0u,0u};
            if (s < SPAT) {
                const uint4* p = (const uint4*)(rpnF + (size_t)s * CR + k0 + kh);
                v0 = p[0]; v1 = p[1];
            }
            *(uint4*)&As[row][kh]     = v0;
            *(uint4*)&As[row][kh + 8] = v1;
        }
        for (int u = t; u < 32 * 14; u += 256) {    // 32 k-rows x 14 groups of 8
            int k  = u / 14;
            int n8 = (u % 14) * 8;
            h8 v = *(const h8*)(w2 + (size_t)(k0 + k) * N2 + n8);
            #pragma unroll
            for (int j = 0; j < 8; ++j) Bs[n8 + j][k] = v[j];
        }
        __syncthreads();

        int mrow = wave * 16 + (lane & 15);
        int kb   = (lane >> 4) * 8;
        h8 alo = *(const h8*)&As[mrow][kb];
        h8 ahi = *(const h8*)&As[mrow][16 + kb];
        v16h a;
        #pragma unroll
        for (int i = 0; i < 8; ++i) { a[i] = alo[i]; a[i + 8] = ahi[i]; }
        int kb2 = (lane >> 4) * 8;
        #pragma unroll
        for (int nt = 0; nt < 7; ++nt) {
            int ncol = nt * 16 + (lane & 15);
            h8 blo = *(const h8*)&Bs[ncol][kb2];
            h8 bhi = *(const h8*)&Bs[ncol][16 + kb2];
            v16h b;
            #pragma unroll
            for (int i = 0; i < 8; ++i) { b[i] = blo[i]; b[i + 8] = bhi[i]; }
            acc[nt] = __builtin_amdgcn_wmma_f32_16x16x32_f16(
                false, a, false, b, (short)0, acc[nt], false, false);
        }
        __syncthreads();
    }

    const int lm = lane & 15;
    const int hi = lane >> 4;
    #pragma unroll
    for (int nt = 0; nt < 7; ++nt) {
        int n = nt * 16 + lm;
        if (n >= N2R) continue;
        float bv = b2[n];
        #pragma unroll
        for (int r = 0; r < 8; ++r) {
            int m = m0 + wave * 16 + r + hi * 8;
            if (m < SPAT) heads[(size_t)m * N2R + n] = acc[nt][r] + bv;
        }
    }
}

// ---------------------------------------------------------------------------
// Proposals: scores (2-way softmax), anchors, bbox_transform_inv + clip
// ---------------------------------------------------------------------------
__global__ void proposals_kernel(const float* __restrict__ heads,
                                 const float* __restrict__ im_info,
                                 float* __restrict__ scores,
                                 float* __restrict__ boxes) {
    int i = blockIdx.x * 256 + threadIdx.x;
    if (i >= NPROP) return;
    int a = i % 9;
    int s = i / 9;
    const float* h = heads + (size_t)s * N2R;
    float bg = h[a], fg = h[9 + a];
    scores[i] = 1.f / (1.f + expf(bg - fg));

    // base anchor a = ratio_idx*3 + scale_idx
    int   ri = a / 3, si = a % 3;
    float ratio = (ri == 0) ? 0.5f : (ri == 1 ? 1.f : 2.f);
    float scale = (si == 0) ? 8.f  : (si == 1 ? 16.f : 32.f);
    float wsr = rintf(sqrtf(256.f / ratio));
    float hsr = rintf(wsr * ratio);
    float wsc = wsr * scale, hsc = hsr * scale;
    int   y = s / IMG_W, x = s - y * IMG_W;
    float shx = x * 16.f, shy = y * 16.f;
    float ax1 = shx + 7.5f - 0.5f * (wsc - 1.f);
    float ay1 = shy + 7.5f - 0.5f * (hsc - 1.f);
    float ax2 = shx + 7.5f + 0.5f * (wsc - 1.f);
    float ay2 = shy + 7.5f + 0.5f * (hsc - 1.f);
    float aw = ax2 - ax1 + 1.f, ah = ay2 - ay1 + 1.f;
    float acx = ax1 + 0.5f * aw, acy = ay1 + 0.5f * ah;

    float d0 = h[18 + a * 4], d1 = h[18 + a * 4 + 1];
    float d2 = h[18 + a * 4 + 2], d3 = h[18 + a * 4 + 3];
    float pcx = d0 * aw + acx, pcy = d1 * ah + acy;
    float pw = expf(d2) * aw,  ph = expf(d3) * ah;
    float imH = im_info[0], imW = im_info[1];
    boxes[i * 4 + 0] = fminf(fmaxf(pcx - 0.5f * pw, 0.f), imW - 1.f);
    boxes[i * 4 + 1] = fminf(fmaxf(pcy - 0.5f * ph, 0.f), imH - 1.f);
    boxes[i * 4 + 2] = fminf(fmaxf(pcx + 0.5f * pw, 0.f), imW - 1.f);
    boxes[i * 4 + 3] = fminf(fmaxf(pcy + 0.5f * ph, 0.f), imH - 1.f);
}

// ---------------------------------------------------------------------------
// top-300 (iterative argmax, destructive on scores) + rois + affine params
// ---------------------------------------------------------------------------
__global__ __launch_bounds__(1024) void topk_kernel(
    float* __restrict__ scores, const float* __restrict__ boxes,
    const float* __restrict__ heads, int* __restrict__ idxbuf,
    float* __restrict__ rois, float* __restrict__ theta,
    float* __restrict__ theta2) {
    __shared__ float sv[1024];
    __shared__ int   si[1024];
    const int t = threadIdx.x;

    for (int it = 0; it < POST; ++it) {
        float best = -1e30f; int bi = NPROP;
        for (int j = t; j < NPROP; j += 1024) {
            float v = scores[j];
            if (v > best || (v == best && j < bi)) { best = v; bi = j; }
        }
        sv[t] = best; si[t] = bi;
        __syncthreads();
        for (int off = 512; off > 0; off >>= 1) {
            if (t < off) {
                if (sv[t + off] > sv[t] ||
                    (sv[t + off] == sv[t] && si[t + off] < si[t])) {
                    sv[t] = sv[t + off]; si[t] = si[t + off];
                }
            }
            __syncthreads();
        }
        if (t == 0) { idxbuf[it] = si[0]; scores[si[0]] = -1e38f; }
        __syncthreads();
    }

    if (t < POST) {
        int j = idxbuf[t];
        float x1 = boxes[j * 4], y1 = boxes[j * 4 + 1];
        float x2 = boxes[j * 4 + 2], y2 = boxes[j * 4 + 3];
        rois[t * 5 + 0] = 0.f;
        rois[t * 5 + 1] = x1; rois[t * 5 + 2] = y1;
        rois[t * 5 + 3] = x2; rois[t * 5 + 4] = y2;
        float fx1 = x1 / 16.f, fy1 = y1 / 16.f, fx2 = x2 / 16.f, fy2 = y2 / 16.f;
        theta[t * 4 + 0] = (fx2 - fx1) / 74.f;                 // t00
        theta[t * 4 + 1] = (fx1 + fx2 - 74.f) / 74.f;          // t02
        theta[t * 4 + 2] = (fy2 - fy1) / 49.f;                 // t11
        theta[t * 4 + 3] = (fy1 + fy2 - 49.f) / 49.f;          // t12
        int s = j / 9, a = j % 9;
        const float* h = heads + (size_t)s * N2R + 54 + a * 6;
        #pragma unroll
        for (int m = 0; m < 6; ++m) theta2[t * 6 + m] = h[m];
    }
}

// ---------------------------------------------------------------------------
// Bilinear sampling (zeros padding, align_corners semantics of reference)
// ---------------------------------------------------------------------------
__device__ __forceinline__ float bilin(const float* __restrict__ im,
                                       float x, float y, int Wd, int Hd) {
    float x0f = floorf(x), y0f = floorf(y);
    int   x0 = (int)x0f,   y0 = (int)y0f;
    float wx1 = x - x0f, wx0 = 1.f - wx1;
    float wy1 = y - y0f, wy0 = 1.f - wy1;
    float acc = 0.f;
    #pragma unroll
    for (int jy = 0; jy < 2; ++jy) {
        #pragma unroll
        for (int jx = 0; jx < 2; ++jx) {
            int xi = x0 + jx, yi = y0 + jy;
            if (xi >= 0 && xi < Wd && yi >= 0 && yi < Hd) {
                float w = (jx ? wx1 : wx0) * (jy ? wy1 : wy0);
                acc += im[yi * Wd + xi] * w;
            }
        }
    }
    return acc;
}

// ---------------------------------------------------------------------------
// ROI affine-grid pooling: 14x14 bilinear crop + 2x2 max -> [300][1024][7][7]
// ---------------------------------------------------------------------------
__global__ void roi_pool_kernel(const float* __restrict__ img,    // [CIN][50][75]
                                const float* __restrict__ theta,  // [300][4]
                                float* __restrict__ pooled) {
    int gid = blockIdx.x * 256 + threadIdx.x;
    if (gid >= POOLE) return;
    int p  = gid % 49;
    int c  = (gid / 49) % 1024;
    int r  = gid / (49 * 1024);
    int px = p % 7, py = p / 7;
    float t00 = theta[r * 4 + 0], t02 = theta[r * 4 + 1];
    float t11 = theta[r * 4 + 2], t12 = theta[r * 4 + 3];
    const float* im = img + (size_t)c * SPAT;
    float best = -1e38f;
    #pragma unroll
    for (int j = 0; j < 2; ++j) {
        #pragma unroll
        for (int i = 0; i < 2; ++i) {
            int sx = 2 * px + i, sy = 2 * py + j;
            float gx = -1.f + 2.f * sx / 13.f;
            float gy = -1.f + 2.f * sy / 13.f;
            float xt = t00 * gx + t02;
            float yt = t11 * gy + t12;
            float xp = (xt + 1.f) * 0.5f * (IMG_W - 1);
            float yp = (yt + 1.f) * 0.5f * (IMG_H - 1);
            best = fmaxf(best, bilin(im, xp, yp, IMG_W, IMG_H));
        }
    }
    pooled[gid] = best;
}

// ---------------------------------------------------------------------------
// Per-ROI spatial transform on pooled 7x7 maps
// ---------------------------------------------------------------------------
__global__ void roi_transform_kernel(const float* __restrict__ pooled,
                                     const float* __restrict__ theta2,
                                     float* __restrict__ out) {
    int gid = blockIdx.x * 256 + threadIdx.x;
    if (gid >= POOLE) return;
    int p  = gid % 49;
    int c  = (gid / 49) % 1024;
    int r  = gid / (49 * 1024);
    int px = p % 7, py = p / 7;
    const float* th = theta2 + r * 6;
    float gx = (px - 3) / 3.f;
    float gy = (py - 3) / 3.f;
    float xt = th[0] * gx + th[1] * gy + th[2];
    float yt = th[3] * gx + th[4] * gy + th[5];
    float xp = (xt + 1.f) * 3.f;
    float yp = (yt + 1.f) * 3.f;
    const float* im = pooled + ((size_t)r * 1024 + c) * 49;
    out[gid] = bilin(im, xp, yp, 7, 7);
}

// ---------------------------------------------------------------------------
// Launcher
// ---------------------------------------------------------------------------
static inline size_t align256(size_t x) { return (x + 255) & ~(size_t)255; }

extern "C" void kernel_launch(void* const* d_in, const int* in_sizes, int n_in,
                              void* d_out, int out_size, void* d_ws, size_t ws_size,
                              hipStream_t stream) {
    const float* head   = (const float*)d_in[0];
    const float* iminfo = (const float*)d_in[2];
    const float* w_rpn  = (const float*)d_in[3];
    const float* b_rpn  = (const float*)d_in[4];
    const float* w_cls  = (const float*)d_in[5];
    const float* b_cls  = (const float*)d_in[6];
    const float* w_bbox = (const float*)d_in[7];
    const float* b_bbox = (const float*)d_in[8];
    const float* w_trans= (const float*)d_in[9];
    const float* b_trans= (const float*)d_in[10];

    char* ws = (char*)d_ws;
    size_t off = 0;
    _Float16* inF   = (_Float16*)(ws + off); off = align256(off + (size_t)SPAT * CIN * 2);
    _Float16* wT    = (_Float16*)(ws + off); off = align256(off + (size_t)9 * CIN * CR * 2);
    _Float16* rpnF  = (_Float16*)(ws + off); off = align256(off + (size_t)SPAT * CR * 2);
    _Float16* w2    = (_Float16*)(ws + off); off = align256(off + (size_t)CR * N2 * 2);
    float*    b2    = (float*)(ws + off);    off = align256(off + (size_t)N2 * 4);
    float*    heads = (float*)(ws + off);    off = align256(off + (size_t)SPAT * N2R * 4);
    float*    scores= (float*)(ws + off);    off = align256(off + (size_t)NPROP * 4);
    float*    boxes = (float*)(ws + off);    off = align256(off + (size_t)NPROP * 4 * 4);
    int*      idxb  = (int*)(ws + off);      off = align256(off + (size_t)POST * 4);
    float*    theta = (float*)(ws + off);    off = align256(off + (size_t)POST * 4 * 4);
    float*    theta2= (float*)(ws + off);    off = align256(off + (size_t)POST * 6 * 4);

    float* rois        = (float*)d_out;                    // [300][5]
    float* pooled      = rois + POST * 5;                  // [300][1024][7][7]
    float* transformed = pooled + POOLE;                   // [300][1024][7][7]

    // 1) layout / dtype prep
    prep_input<<<(SPAT * CIN + 255) / 256, 256, 0, stream>>>(head, inF);
    prep_wrpn <<<(9 * CIN * CR + 255) / 256, 256, 0, stream>>>(w_rpn, wT);
    prep_w2   <<<(CR * N2 + 255) / 256, 256, 0, stream>>>(w_cls, w_bbox, w_trans,
                                                          b_cls, b_bbox, b_trans, w2, b2);
    // 2) 3x3 conv via WMMA (M=3750 -> 30 M-blocks of 128; N=512 -> 8 N-blocks of 64)
    rpn_conv_gemm<<<dim3(30, 8), 256, 0, stream>>>(inF, wT, b_rpn, rpnF);
    // 3) fused 1x1 heads via WMMA
    head_gemm<<<30, 256, 0, stream>>>(rpnF, w2, b2, heads);
    // 4) scores + boxes
    proposals_kernel<<<(NPROP + 255) / 256, 256, 0, stream>>>(heads, iminfo, scores, boxes);
    // 5) top-300 + rois + affine params
    topk_kernel<<<1, 1024, 0, stream>>>(scores, boxes, heads, idxb, rois, theta, theta2);
    // 6) affine-grid ROI pooling (14x14 bilinear + 2x2 max)
    roi_pool_kernel<<<(POOLE + 255) / 256, 256, 0, stream>>>(head, theta, pooled);
    // 7) per-ROI spatial transform of pooled maps
    roi_transform_kernel<<<(POOLE + 255) / 256, 256, 0, stream>>>(pooled, theta2, transformed);
}
